// SlotAttention_52767968199203
// MI455X (gfx1250) — matrix operand order repaired and checked by reading, not compiled, not run on hardware
//
#include <hip/hip_runtime.h>
#include <hip/hip_bf16.h>

typedef __attribute__((ext_vector_type(16))) _Float16 v16h;
typedef __attribute__((ext_vector_type(8)))  _Float16 v8h;
typedef __attribute__((ext_vector_type(8)))  float    v8f;

// Problem constants
#define NB    64
#define NPOS  4096
#define NFEAT 256
#define NKVQ  256
#define NHEAD 4
#define NDPH  64
#define NSLOT 8

__device__ __forceinline__ v8f zero8() {
  v8f z = {0.f,0.f,0.f,0.f,0.f,0.f,0.f,0.f};
  return z;
}

// A-fragment (16x32 f16): per-lane two contiguous 16B chunks at +0 and +8 halves.
__device__ __forceinline__ v16h load_afrag(const _Float16* p) {
  const v8h lo = *(const v8h*)(p);
  const v8h hi = *(const v8h*)(p + 8);
  v16h r;
#pragma unroll
  for (int e = 0; e < 8; ++e) { r[e] = lo[e]; r[e + 8] = hi[e]; }
  return r;
}

__device__ __forceinline__ v8f wmma16(v16h a, v16h b, v8f c) {
  return __builtin_amdgcn_wmma_f32_16x16x32_f16(false, a, false, b, (short)0, c,
                                                false, false);
}

// ---------------------------------------------------------------------------
// Kw: pack [Wk;Wv] into f16 [512,256]
// ---------------------------------------------------------------------------
__global__ __launch_bounds__(256) void k_cvt_w(const float* __restrict__ Wk,
                                               const float* __restrict__ Wv,
                                               _Float16* __restrict__ Wh) {
  const int idx = blockIdx.x * 256 + threadIdx.x;  // 0..131071
  const float v = (idx < 65536) ? Wk[idx] : Wv[idx - 65536];
  Wh[idx] = (_Float16)v;
}

// ---------------------------------------------------------------------------
// K1: fused input LayerNorm + K/V projection GEMM (WMMA f16, f32 accum).
// Block = 64 rows x 512 cols, K=256.  8 waves: 4 m-tiles x 2 n-groups(16 tiles).
// x tile is read from global exactly once (register-cached across the stats
// barrier).  B-fragments are loaded in groups of 8 ahead of the 8 WMMAs so
// loads and matrix ops overlap instead of ping-ponging on s_wait_loadcnt 0.
// k_s layout [b,h,j,d] f16 ; v_t layout [b,h,d,j] f16.
// ---------------------------------------------------------------------------
__global__ __launch_bounds__(256) void k_ln_kv(const float* __restrict__ x,
                                               const float* __restrict__ lnw,
                                               const float* __restrict__ lnb,
                                               const _Float16* __restrict__ Wh,
                                               _Float16* __restrict__ k_s,
                                               _Float16* __restrict__ v_t) {
  __shared__ _Float16 Ald[64 * 256];
  __shared__ float red[256], red2[256];
  __shared__ float mu_s[64], rs_s[64];
  const int tid = threadIdx.x;
  const size_t row0 = (size_t)blockIdx.x * 64;

  // LN: single global read; raw values live in registers across the barrier.
  {
    const int r = tid >> 2, seg = tid & 3;
    const float4* p = (const float4*)(x + (row0 + r) * 256 + seg * 64);
    float4 vals[16];
    float s = 0.f, sq = 0.f;
#pragma unroll
    for (int c = 0; c < 16; ++c) {
      const float4 f = p[c];
      vals[c] = f;
      s += f.x + f.y + f.z + f.w;
      sq += f.x * f.x + f.y * f.y + f.z * f.z + f.w * f.w;
    }
    red[tid] = s; red2[tid] = sq;
    __syncthreads();
    if (tid < 64) {
      const float ts  = red[tid*4] + red[tid*4+1] + red[tid*4+2] + red[tid*4+3];
      const float tsq = red2[tid*4] + red2[tid*4+1] + red2[tid*4+2] + red2[tid*4+3];
      const float mu = ts * (1.f / 256.f);
      const float var = tsq * (1.f / 256.f) - mu * mu;
      mu_s[tid] = mu;
      rs_s[tid] = rsqrtf(var + 1e-5f);
    }
    __syncthreads();
    const float mu = mu_s[r], rs = rs_s[r];
    _Float16* ap = &Ald[r * 256 + seg * 64];
    const float* wv = lnw + seg * 64;
    const float* bv = lnb + seg * 64;
#pragma unroll
    for (int c = 0; c < 16; ++c) {
      ap[4*c+0] = (_Float16)((vals[c].x - mu) * rs * wv[4*c+0] + bv[4*c+0]);
      ap[4*c+1] = (_Float16)((vals[c].y - mu) * rs * wv[4*c+1] + bv[4*c+1]);
      ap[4*c+2] = (_Float16)((vals[c].z - mu) * rs * wv[4*c+2] + bv[4*c+2]);
      ap[4*c+3] = (_Float16)((vals[c].w - mu) * rs * wv[4*c+3] + bv[4*c+3]);
    }
  }
  __syncthreads();

  // GEMM
  const int lane = tid & 31, wid = tid >> 5;
  const int m16 = wid & 3, ng = wid >> 2;
  const int kbA = (lane & 16) ? 16 : 0;
  const int kbB = (lane & 16) ? 16 : 0;
  v8f acc[16];
#pragma unroll
  for (int t = 0; t < 16; ++t) acc[t] = zero8();

  for (int kc = 0; kc < 8; ++kc) {
    const v16h a = load_afrag(&Ald[(m16 * 16 + (lane & 15)) * 256 + kc * 32 + kbA]);
#pragma unroll
    for (int g = 0; g < 2; ++g) {
      v16h bf[8];
#pragma unroll
      for (int q = 0; q < 8; ++q) {
        const int n = ng * 256 + (g * 8 + q) * 16 + (lane & 15);
        bf[q] = *(const v16h*)(Wh + (size_t)n * 256 + kc * 32 + kbB);
      }
#pragma unroll
      for (int q = 0; q < 8; ++q) acc[g * 8 + q] = wmma16(a, bf[q], acc[g * 8 + q]);
    }
  }

  // Epilogue: write k_s / v_t (transposed)
  const int mtop = (lane & 16) ? 8 : 0;
#pragma unroll
  for (int nt = 0; nt < 16; ++nt) {
    const int n = ng * 256 + nt * 16 + (lane & 15);
#pragma unroll
    for (int r = 0; r < 8; ++r) {
      const size_t grow = row0 + (size_t)(m16 * 16 + mtop + r);
      const int bb = (int)(grow >> 12);
      const int jj = (int)(grow & 4095);
      const float val = acc[nt][r];
      if (n < 256) {
        const int h = n >> 6, d = n & 63;
        k_s[(((size_t)(bb * NHEAD + h) * NPOS) + jj) * NDPH + d] = (_Float16)val;
      } else {
        const int n2 = n - 256;
        const int h = n2 >> 6, d = n2 & 63;
        v_t[(((size_t)(bb * NHEAD + h) * NDPH) + d) * NPOS + jj] = (_Float16)val;
      }
    }
  }
}

// ---------------------------------------------------------------------------
// K2: slot LayerNorm + q projection (scaled).  q_s layout [b,h,i,d] f16.
// ---------------------------------------------------------------------------
__global__ __launch_bounds__(256) void k_slot_q(const float* __restrict__ slots,
                                                const float* __restrict__ lnw,
                                                const float* __restrict__ lnb,
                                                const float* __restrict__ Wq,
                                                _Float16* __restrict__ q_s) {
  __shared__ float red[256], redq[256], sn[256];
  const int tid = threadIdx.x;
  const int row = blockIdx.x;  // (b*8 + i)
  const float v = slots[(size_t)row * 256 + tid];
  red[tid] = v; redq[tid] = v * v;
  __syncthreads();
  for (int off = 128; off > 0; off >>= 1) {
    if (tid < off) { red[tid] += red[tid + off]; redq[tid] += redq[tid + off]; }
    __syncthreads();
  }
  const float mu = red[0] * (1.f / 256.f);
  const float var = redq[0] * (1.f / 256.f) - mu * mu;
  const float rs = rsqrtf(var + 1e-5f);
  sn[tid] = (v - mu) * rs * lnw[tid] + lnb[tid];
  __syncthreads();
  const float* wrow = Wq + (size_t)tid * 256;
  float acc = 0.f;
  for (int c = 0; c < 256; ++c) acc = fmaf(sn[c], wrow[c], acc);
  acc *= 0.125f;  // SCALE = 64^-0.5
  const int b = row >> 3, i = row & 7;
  const int h = tid >> 6, d = tid & 63;
  q_s[((size_t)((b * NHEAD + h) * NSLOT + i)) * NDPH + d] = (_Float16)acc;
}

// ---------------------------------------------------------------------------
// K3a: dots = q @ k^T per (b,h) via WMMA. M=16 (8 slots + pad), K=64, N=4096.
// Software-pipelined: next tile's k fragments are loaded before this tile's
// WMMAs so global latency is hidden behind the matrix ops.
// dots layout [b, j, 32] f32 (32 = h*8 + i) for softmax over sloth axis.
// ---------------------------------------------------------------------------
__global__ __launch_bounds__(256) void k_dots(const _Float16* __restrict__ q_s,
                                              const _Float16* __restrict__ k_s,
                                              float* __restrict__ dots) {
  const int tid = threadIdx.x, lane = tid & 31, wid = tid >> 5;
  const int bb = blockIdx.x >> 2, h = blockIdx.x & 3;
  const int kbA = (lane & 16) ? 16 : 0;
  const int kbB = (lane & 16) ? 16 : 0;
  const _Float16* qrow =
      q_s + ((size_t)((bb * NHEAD + h) * NSLOT + (lane & 7))) * NDPH;
  const v16h a0 = load_afrag(qrow + 0 + kbA);
  const v16h a1 = load_afrag(qrow + 32 + kbA);
  const _Float16* kbase = k_s + (size_t)(bb * NHEAD + h) * NPOS * NDPH;

  const _Float16* krow = kbase + (size_t)(wid * 512 + (lane & 15)) * NDPH;
  v16h b0 = *(const v16h*)(krow + 0 + kbB);
  v16h b1 = *(const v16h*)(krow + 32 + kbB);
  for (int t = 0; t < 32; ++t) {
    const int j0 = (wid * 32 + t) * 16;
    const v16h c0 = b0, c1 = b1;
    if (t < 31) {
      const _Float16* krn = kbase + (size_t)(j0 + 16 + (lane & 15)) * NDPH;
      b0 = *(const v16h*)(krn + 0 + kbB);
      b1 = *(const v16h*)(krn + 32 + kbB);
    }
    v8f acc = zero8();
    acc = wmma16(a0, c0, acc);
    acc = wmma16(a1, c1, acc);
    if (lane < 16) {
      float* dp = dots + ((size_t)bb * NPOS + j0 + lane) * 32 + h * 8;
#pragma unroll
      for (int r = 0; r < 8; ++r) dp[r] = acc[r];
    }
  }
}

// ---------------------------------------------------------------------------
// K3b: softmax over 32 sloth values per (b,j); writes attn f16 rows
// [(b,i,h), j], and attn_before mean-over-heads to d_out stack (+ last iter).
// ---------------------------------------------------------------------------
__global__ __launch_bounds__(256) void k_softmax(const float* __restrict__ dots,
                                                 _Float16* __restrict__ attn,
                                                 float* __restrict__ out,
                                                 int iter) {
  const int p = blockIdx.x * 256 + threadIdx.x;
  const int b = p >> 12, j = p & 4095;
  float d[32];
  const float4* dp = (const float4*)(dots + (size_t)p * 32);
#pragma unroll
  for (int t = 0; t < 8; ++t) {
    const float4 f = dp[t];
    d[4*t] = f.x; d[4*t+1] = f.y; d[4*t+2] = f.z; d[4*t+3] = f.w;
  }
  float m = d[0];
#pragma unroll
  for (int s = 1; s < 32; ++s) m = fmaxf(m, d[s]);
  float sum = 0.f;
#pragma unroll
  for (int s = 0; s < 32; ++s) { d[s] = __expf(d[s] - m); sum += d[s]; }
  const float inv = 1.f / sum;
#pragma unroll
  for (int s = 0; s < 32; ++s) {
    d[s] *= inv;
    const int i = s & 7, h = s >> 3;
    attn[((size_t)((b * NSLOT + i) * NHEAD + h)) * NPOS + j] = (_Float16)d[s];
  }
  float* stack = out + 2228224 + ((size_t)(iter * NB + b) * NSLOT) * NPOS;
  float* last  = out + 131072  + ((size_t)(b * NSLOT)) * NPOS;
#pragma unroll
  for (int i = 0; i < 8; ++i) {
    const float mean = 0.25f * (d[i] + d[8 + i] + d[16 + i] + d[24 + i]);
    stack[(size_t)i * NPOS + j] = mean;
    if (iter == 2) last[(size_t)i * NPOS + j] = mean;
  }
}

// ---------------------------------------------------------------------------
// K3c: per-row sums of attn (for slot-wise renormalization)
// ---------------------------------------------------------------------------
__global__ __launch_bounds__(256) void k_rowsum(const _Float16* __restrict__ attn,
                                                float* __restrict__ sums) {
  __shared__ float red[256];
  const int tid = threadIdx.x;
  const _Float16* row = attn + (size_t)blockIdx.x * NPOS;
  float s = 0.f;
  for (int k = 0; k < 16; ++k) s += (float)row[tid + k * 256];
  red[tid] = s;
  __syncthreads();
  for (int off = 128; off > 0; off >>= 1) {
    if (tid < off) red[tid] += red[tid + off];
    __syncthreads();
  }
  if (tid == 0) sums[blockIdx.x] = red[0];
}

// ---------------------------------------------------------------------------
// K4: updates = attn @ v per (b,h) via WMMA over K=4096, then renormalize by
// (rowsum + eps).  Software-pipelined A/B fragment loads; deterministic
// fixed-order cross-wave LDS reduction.
// ---------------------------------------------------------------------------
__global__ __launch_bounds__(256) void k_updates(const _Float16* __restrict__ attn,
                                                 const _Float16* __restrict__ v_t,
                                                 const float* __restrict__ sums,
                                                 float* __restrict__ updates) {
  __shared__ float part[8][512];  // per-wave 8x64 partial tiles
  const int tid = threadIdx.x, lane = tid & 31, wid = tid >> 5;
  const int bb = blockIdx.x >> 2, h = blockIdx.x & 3;
  const int kbA = (lane & 16) ? 16 : 0;
  const int kbB = (lane & 16) ? 16 : 0;
  const _Float16* arow =
      attn + ((size_t)((bb * NSLOT + (lane & 7)) * NHEAD + h)) * NPOS;
  const _Float16* vbase = v_t + (size_t)(bb * NHEAD + h) * NDPH * NPOS +
                          (size_t)(lane & 15) * NPOS;
  v8f acc[4] = {zero8(), zero8(), zero8(), zero8()};

  const int jb0 = wid * 512;
  v16h a = load_afrag(arow + jb0 + kbA);
  v16h bf[4];
#pragma unroll
  for (int nt = 0; nt < 4; ++nt)
    bf[nt] = *(const v16h*)(vbase + (size_t)(nt * 16) * NPOS + jb0 + kbB);

  for (int t = 0; t < 16; ++t) {
    const v16h ca = a;
    v16h cb[4];
#pragma unroll
    for (int nt = 0; nt < 4; ++nt) cb[nt] = bf[nt];
    if (t < 15) {
      const int jb = jb0 + (t + 1) * 32;
      a = load_afrag(arow + jb + kbA);
#pragma unroll
      for (int nt = 0; nt < 4; ++nt)
        bf[nt] = *(const v16h*)(vbase + (size_t)(nt * 16) * NPOS + jb + kbB);
    }
#pragma unroll
    for (int nt = 0; nt < 4; ++nt) acc[nt] = wmma16(ca, cb[nt], acc[nt]);
  }

  if (lane < 16) {
#pragma unroll
    for (int nt = 0; nt < 4; ++nt)
#pragma unroll
      for (int r = 0; r < 8; ++r)
        part[wid][r * 64 + nt * 16 + lane] = acc[nt][r];
  }
  __syncthreads();
  for (int idx = tid; idx < 512; idx += 256) {
    float s = 0.f;
#pragma unroll
    for (int w = 0; w < 8; ++w) s += part[w][idx];  // fixed order
    const int i = idx >> 6, d = idx & 63;
    const float rs = sums[(bb * NSLOT + i) * NHEAD + h];
    updates[((size_t)(bb * NSLOT + i)) * 256 + h * 64 + d] = s / (rs + 1e-8f);
  }
}

// ---------------------------------------------------------------------------
// K5: GRU cell (torch semantics), slots updated in place; last iter -> d_out.
// ---------------------------------------------------------------------------
__global__ __launch_bounds__(256) void k_gru(const float* __restrict__ updates,
                                             float* __restrict__ slots,
                                             const float* __restrict__ wih,
                                             const float* __restrict__ whh,
                                             const float* __restrict__ bih,
                                             const float* __restrict__ bhh,
                                             float* __restrict__ out, int iter) {
  __shared__ float u[256], hprev[256];
  const int tid = threadIdx.x;
  const size_t row = blockIdx.x;
  u[tid] = updates[row * 256 + tid];
  hprev[tid] = slots[row * 256 + tid];
  __syncthreads();
  float gx[3], gh[3];
#pragma unroll
  for (int g = 0; g < 3; ++g) {
    const float* wr = wih + (size_t)(g * 256 + tid) * 256;
    const float* hr = whh + (size_t)(g * 256 + tid) * 256;
    float ax = bih[g * 256 + tid], ah = bhh[g * 256 + tid];
    for (int c = 0; c < 256; ++c) {
      ax = fmaf(u[c], wr[c], ax);
      ah = fmaf(hprev[c], hr[c], ah);
    }
    gx[g] = ax; gh[g] = ah;
  }
  const float r = 1.f / (1.f + __expf(-(gx[0] + gh[0])));
  const float z = 1.f / (1.f + __expf(-(gx[1] + gh[1])));
  const float n = tanhf(gx[2] + r * gh[2]);
  const float hnew = (1.f - z) * n + z * hprev[tid];
  slots[row * 256 + tid] = hnew;
  if (iter == 2) out[row * 256 + tid] = hnew;
}

// ---------------------------------------------------------------------------
extern "C" void kernel_launch(void* const* d_in, const int* in_sizes, int n_in,
                              void* d_out, int out_size, void* d_ws,
                              size_t ws_size, hipStream_t stream) {
  (void)in_sizes; (void)n_in; (void)out_size; (void)ws_size;
  const float* inputs    = (const float*)d_in[0];
  const float* cond      = (const float*)d_in[1];
  const float* ln_in_w   = (const float*)d_in[2];
  const float* ln_in_b   = (const float*)d_in[3];
  const float* ln_slot_w = (const float*)d_in[4];
  const float* ln_slot_b = (const float*)d_in[5];
  const float* Wq        = (const float*)d_in[6];
  const float* Wk        = (const float*)d_in[7];
  const float* Wv        = (const float*)d_in[8];
  const float* gwih      = (const float*)d_in[9];
  const float* gwhh      = (const float*)d_in[10];
  const float* gbih      = (const float*)d_in[11];
  const float* gbhh      = (const float*)d_in[12];
  float* out = (float*)d_out;

  char* ws = (char*)d_ws;
  _Float16* Wh   = (_Float16*)ws; ws += 262144;     // [512,256] f16
  _Float16* k_s  = (_Float16*)ws; ws += 134217728;  // [b,h,j,d] f16
  _Float16* v_t  = (_Float16*)ws; ws += 134217728;  // [b,h,d,j] f16
  float*    dots = (float*)ws;    ws += 33554432;   // [b,j,32] f32
  _Float16* attn = (_Float16*)ws; ws += 16777216;   // [(b,i,h), j] f16
  float*    sums = (float*)ws;    ws += 8192;       // [2048]
  float*    upd  = (float*)ws;    ws += 524288;     // [512,256] f32
  _Float16* q_s  = (_Float16*)ws; ws += 262144;     // [b,h,i,d] f16
  float*    slots= (float*)ws;    ws += 524288;     // [512,256] f32

  hipMemcpyAsync(slots, cond, 512 * 256 * sizeof(float),
                 hipMemcpyDeviceToDevice, stream);
  k_cvt_w<<<512, 256, 0, stream>>>(Wk, Wv, Wh);
  k_ln_kv<<<4096, 256, 0, stream>>>(inputs, ln_in_w, ln_in_b, Wh, k_s, v_t);
  for (int iter = 0; iter < 3; ++iter) {
    k_slot_q<<<512, 256, 0, stream>>>(slots, ln_slot_w, ln_slot_b, Wq, q_s);
    k_dots<<<256, 256, 0, stream>>>(q_s, k_s, dots);
    k_softmax<<<1024, 256, 0, stream>>>(dots, attn, out, iter);
    k_rowsum<<<2048, 256, 0, stream>>>(attn, sums);
    k_updates<<<256, 256, 0, stream>>>(attn, v_t, sums, upd);
    k_gru<<<512, 256, 0, stream>>>(upd, slots, gwih, gwhh, gbih, gbhh, out, iter);
  }
}
